// GINCriticNetwork_29703993819801
// MI455X (gfx1250) — matrix operand order repaired and checked by reading, not compiled, not run on hardware
//
#include <hip/hip_runtime.h>
#include <hip/hip_bf16.h>
#include <math.h>

typedef __attribute__((ext_vector_type(16))) _Float16 v16h;
typedef __attribute__((ext_vector_type(8)))  float    v8f;

// ---- problem constants ----
#define BATCH 4096
#define NAG   64
#define HID   128
#define FEAT  27      // 7 orig + 20 lidar
#define LIDAR 20
#define PROC  20
#define FOVC  (0.35f * 3.14159265358979323846f)
#define MAXD  10.0f

// ---- f16 weight workspace offsets (in _Float16 elements) ----
// fragment-swizzled: frag f = kT*NT+nT; element ((f*32+lane)*16 + j)
#define FR(KT,NT) ((KT)*(NT)*32*16)
static const size_t OFF_WL1 = 0;                          // KT=1 NT=8
static const size_t OFF_WL2 = OFF_WL1 + FR(1,8);          // KT=4 NT=2
static const size_t OFF_W11 = OFF_WL2 + FR(4,2);          // KT=1 NT=8
static const size_t OFF_W12 = OFF_W11 + FR(1,8);          // KT=4 NT=8
static const size_t OFF_W21 = OFF_W12 + FR(4,8);
static const size_t OFF_W22 = OFF_W21 + FR(4,8);
static const size_t OFF_WV  = OFF_W22 + FR(4,8);          // 2 layers each below
static const size_t OFF_WO  = OFF_WV  + 2*FR(4,8);
static const size_t OFF_WF1 = OFF_WO  + 2*FR(4,8);
static const size_t OFF_WF2 = OFF_WF1 + 2*FR(4,8);

// =====================================================================
// Weight conversion: f32 (N,K) row-major -> fragment-swizzled f16
// B-fragment layout (16x16x32 f16 WMMA, wave32):
//   n = nT*16 + (lane&15),  k = kT*32 + ((lane>>4)&1)*16 + j  (j=0..15)
// =====================================================================
__global__ void convert_w(const float* __restrict__ W, _Float16* __restrict__ dst,
                          int K, int N, int KT, int NT) {
  int idx = blockIdx.x * blockDim.x + threadIdx.x;     // one thread per (frag,lane)
  int total = KT * NT * 32;
  if (idx >= total) return;
  int lane = idx & 31;
  int f    = idx >> 5;
  int nT   = f % NT;
  int n  = nT * 16 + (lane & 15);
  int kb = (f / NT) * 32 + ((lane >> 4) & 1) * 16;
  _Float16* d = dst + (size_t)idx * 16;
#pragma unroll
  for (int j = 0; j < 16; ++j) {
    int k = kb + j;
    float v = (k < K && n < N) ? W[(size_t)n * K + k] : 0.0f;
    d[j] = (_Float16)v;
  }
}

// =====================================================================
// Fragment loaders (all LDS pointers come straight from __shared__ arrays
// at every call site -> infer-address-spaces gives ds_load/ds_store)
// A (16x32 f16): m = lane&15 ; K(j) = 16*(lane>>4) + j
// =====================================================================
__device__ __forceinline__ v16h ldA_f16(const _Float16* s, int ld, int mT, int kT, int lane) {
  v16h a;
  const _Float16* r = s + (size_t)(mT * 16 + (lane & 15)) * ld + kT * 32;
  int h8 = ((lane >> 4) & 1) * 16;
#pragma unroll
  for (int j = 0; j < 8; ++j) { a[j] = r[h8 + j]; a[j + 8] = r[8 + h8 + j]; }
  return a;
}
__device__ __forceinline__ v16h ldA_f32(const float* s, int ld, int mT, int kT, int lane) {
  v16h a;
  const float* r = s + (size_t)(mT * 16 + (lane & 15)) * ld + kT * 32;
  int h8 = ((lane >> 4) & 1) * 16;
#pragma unroll
  for (int j = 0; j < 8; ++j) { a[j] = (_Float16)r[h8 + j]; a[j + 8] = (_Float16)r[8 + h8 + j]; }
  return a;
}
// A = transpose(src): A[m][k] = src[k*ld + m]   (mask aggregation)
__device__ __forceinline__ v16h ldA_tr(const _Float16* s, int ld, int mT, int kT, int lane) {
  v16h a;
  int m  = mT * 16 + (lane & 15);
  int kb = kT * 32 + ((lane >> 4) & 1) * 16;
#pragma unroll
  for (int j = 0; j < 8; ++j) {
    a[j]     = s[(size_t)(kb + j) * ld + m];
    a[j + 8] = s[(size_t)(kb + 8 + j) * ld + m];
  }
  return a;
}
// B (32x16 f16) from LDS row-major: n = lane&15 ; k = kT*32 + ((lane>>4)&1)*16 + j
__device__ __forceinline__ v16h ldB_lds(const _Float16* s, int ld, int kT, int nT, int lane) {
  v16h b;
  int n  = nT * 16 + (lane & 15);
  int kb = kT * 32 + ((lane >> 4) & 1) * 16;
#pragma unroll
  for (int j = 0; j < 16; ++j) b[j] = s[(size_t)(kb + j) * ld + n];
  return b;
}
// B from fragment-swizzled global f16 weights: one contiguous 32B load/lane
__device__ __forceinline__ v16h ldB_w(const _Float16* w, int NT, int kT, int nT, int lane) {
  return ((const v16h*)w)[(size_t)(kT * NT + nT) * 32 + lane];
}

// =====================================================================
// Compile-time configured GEMM over a 64xN tile:
//   out = act(64x(KT*32)) @ Wswz + bias (+C) [relu]
// 8 waves, 4*NT tiles, each wave owns whole tiles (EXEC all-ones at WMMA).
// Dual accumulators (even/odd kT) break the dependent WMMA chain.
// =====================================================================
template<bool AF16, int KT, int NT, int LDA, bool RELU, bool HASC, int LDC,
         bool OUTF, int LDO, int NVALID, int COLOFF>
__device__ __forceinline__ void gemm_w(
    const _Float16* aH, const float* aF,
    const _Float16* wf, const float* __restrict__ bias,
    const float* cF, float* oF, _Float16* oH)
{
  const int tid = threadIdx.x, wave = tid >> 5, lane = tid & 31;
  const int n0 = lane & 15;
  const int hi = (lane >> 4) & 1;
  for (int t = wave; t < 4 * NT; t += 8) {
    const int mT = t / NT, nT = t % NT;
    const int n  = nT * 16 + n0;
    const int mb = mT * 16 + hi * 8;
    v8f c0, c1;
    if constexpr (HASC) {
#pragma unroll
      for (int r = 0; r < 8; ++r) c0[r] = cF[(size_t)(mb + r) * LDC + n];
    } else {
#pragma unroll
      for (int r = 0; r < 8; ++r) c0[r] = 0.0f;
    }
#pragma unroll
    for (int r = 0; r < 8; ++r) c1[r] = 0.0f;

#pragma unroll
    for (int k2 = 0; k2 < KT; k2 += 2) {
      v16h a0;
      if constexpr (AF16) a0 = ldA_f16(aH, LDA, mT, k2, lane);
      else                a0 = ldA_f32(aF, LDA, mT, k2, lane);
      v16h b0 = ldB_w(wf, NT, k2, nT, lane);
      c0 = __builtin_amdgcn_wmma_f32_16x16x32_f16(false, a0, false, b0, (short)0, c0, false, false);
      if constexpr (KT > 1) {
        v16h a1;
        if constexpr (AF16) a1 = ldA_f16(aH, LDA, mT, k2 + 1, lane);
        else                a1 = ldA_f32(aF, LDA, mT, k2 + 1, lane);
        v16h b1 = ldB_w(wf, NT, k2 + 1, nT, lane);
        c1 = __builtin_amdgcn_wmma_f32_16x16x32_f16(false, a1, false, b1, (short)0, c1, false, false);
      }
    }
    float bv = (n < NVALID) ? bias[n] : 0.0f;
#pragma unroll
    for (int r = 0; r < 8; ++r) {
      float v = c0[r] + bv;
      if constexpr (KT > 1) v += c1[r];
      if constexpr (RELU) v = fmaxf(v, 0.0f);
      if (n < NVALID) {
        if constexpr (OUTF) oF[(size_t)(mb + r) * LDO + COLOFF + n] = v;
        else                oH[(size_t)(mb + r) * LDO + COLOFF + n] = (_Float16)v;
      }
    }
  }
}

// Aggregation: out = mask^T(64x64) @ Bsrc(64xN) + C  (mask/B/C f16 in LDS)
template<int NT, int LDB, int LDC, bool OUTF, int LDO>
__device__ __forceinline__ void gemm_aggr(
    const _Float16* msk, const _Float16* Bsrc,
    const _Float16* cH, float* oF, _Float16* oH)
{
  const int tid = threadIdx.x, wave = tid >> 5, lane = tid & 31;
  const int n0 = lane & 15;
  const int hi = (lane >> 4) & 1;
  for (int t = wave; t < 4 * NT; t += 8) {
    const int mT = t / NT, nT = t % NT;
    const int n  = nT * 16 + n0;
    const int mb = mT * 16 + hi * 8;
    v8f c0, c1;
#pragma unroll
    for (int r = 0; r < 8; ++r) { c0[r] = (float)cH[(size_t)(mb + r) * LDC + n]; c1[r] = 0.0f; }
    {
      v16h a0 = ldA_tr(msk, NAG, mT, 0, lane);
      v16h b0 = ldB_lds(Bsrc, LDB, 0, nT, lane);
      c0 = __builtin_amdgcn_wmma_f32_16x16x32_f16(false, a0, false, b0, (short)0, c0, false, false);
      v16h a1 = ldA_tr(msk, NAG, mT, 1, lane);
      v16h b1 = ldB_lds(Bsrc, LDB, 1, nT, lane);
      c1 = __builtin_amdgcn_wmma_f32_16x16x32_f16(false, a1, false, b1, (short)0, c1, false, false);
    }
#pragma unroll
    for (int r = 0; r < 8; ++r) {
      float v = c0[r] + c1[r];
      if constexpr (OUTF) oF[(size_t)(mb + r) * LDO + n] = v;
      else                oH[(size_t)(mb + r) * LDO + n] = (_Float16)v;
    }
  }
}

__device__ __forceinline__ void layernorm(float* Sb, const float* __restrict__ g,
                                          const float* __restrict__ bta,
                                          float* st, int tid) {
  if (tid < NAG) {
    const float* r = Sb + (size_t)tid * HID;
    float s = 0.f, s2 = 0.f;
#pragma unroll 4
    for (int k = 0; k < HID; ++k) { float v = r[k]; s += v; s2 += v * v; }
    float mean = s * (1.0f / HID);
    float var  = s2 * (1.0f / HID) - mean * mean;
    st[tid] = mean;
    st[NAG + tid] = rsqrtf(var + 1e-5f);
  }
  __syncthreads();
  for (int idx = tid; idx < NAG * HID; idx += 256) {
    int r = idx >> 7, c = idx & 127;
    float v = (Sb[idx] - st[r]) * st[NAG + r];
    Sb[idx] = v * g[c] + bta[c];
  }
}

// =====================================================================
// Fused per-batch kernel: one block (8 wave32) per graph of 64 agents
// =====================================================================
__global__ void __launch_bounds__(256)
gin_critic_fused(const float* __restrict__ data,
                 const float* __restrict__ b_l1, const float* __restrict__ b_l2,
                 const float* __restrict__ b11,  const float* __restrict__ b12,
                 const float* __restrict__ b21,  const float* __restrict__ b22,
                 const float* __restrict__ bqkv, const float* __restrict__ bo,
                 const float* __restrict__ ln1g, const float* __restrict__ ln1b,
                 const float* __restrict__ bf1,  const float* __restrict__ bf2,
                 const float* __restrict__ ln2g, const float* __restrict__ ln2b,
                 const float* __restrict__ Wfc,  const float* __restrict__ bfc,
                 const _Float16* __restrict__ ws, float* __restrict__ out)
{
  __shared__ float    S[NAG * HID];       // 32 KB: f32 residual/LN stream
  __shared__ _Float16 Hh[NAG * HID];      // 16 KB: f16 activation ping buffer
  __shared__ _Float16 Mh[NAG * NAG];      //  8 KB: mask (also lidar scratch)
  __shared__ _Float16 xh[NAG * 32];       //  4 KB: input features (27 padded to 32)
  __shared__ float    stats[2 * NAG];

  const int tid = threadIdx.x;
  const int b   = blockIdx.x;
  const float* dptr = data + (size_t)b * NAG * FEAT;

  // P0: orig features -> xh cols 0..6 (pad 0), lidar -> Mh scratch (ld 32)
  for (int idx = tid; idx < NAG * 32; idx += 256) {
    int i = idx >> 5, c = idx & 31;
    xh[idx] = (_Float16)((c < 7) ? dptr[i * FEAT + c] : 0.0f);
    Mh[idx] = (_Float16)((c < LIDAR) ? dptr[i * FEAT + 7 + c] : 0.0f);
  }
  __syncthreads();

  // P1: h1 = relu(lidar @ W_l1^T + b_l1)  (K pad 20->32) -> Hh f16
  gemm_w<true, 1, 8, 32, true, false, 0, false, HID, HID, 0>(
      Mh, nullptr, ws + OFF_WL1, b_l1, nullptr, nullptr, Hh);
  __syncthreads();

  // P2: lf = relu(h1 @ W_l2^T + b_l2) -> xh cols 7..26  (N pad 20->32)
  gemm_w<true, 4, 2, HID, true, false, 0, false, 32, PROC, 7>(
      Hh, nullptr, ws + OFF_WL2, b_l2, nullptr, nullptr, xh);
  __syncthreads();

  // P3: adjacency mask from positions (xh cols 0,1) -> Mh (64x64)
  for (int idx = tid; idx < NAG * NAG; idx += 256) {
    int i = idx >> 6, j = idx & 63;
    float dx = (float)xh[i * 32 + 0] - (float)xh[j * 32 + 0];
    float dy = (float)xh[i * 32 + 1] - (float)xh[j * 32 + 1];
    float dist = sqrtf(dx * dx + dy * dy);
    float ang  = atan2f(dy, dx);
    bool ok = (i != j) && (fabsf(ang) <= FOVC) && (dist <= MAXD);
    Mh[idx] = ok ? (_Float16)1.0f : (_Float16)0.0f;
  }
  __syncthreads();

  // P4: x + m^T@x  -> Hh (64x32 region, ld 32)   [C = x folded into WMMA]
  gemm_aggr<2, 32, 32, false, 32>(Mh, xh, xh, nullptr, Hh);
  __syncthreads();

  // P5: t1 = relu((x+aggr) @ W11^T + b11)  (K pad 27->32) -> S f32
  gemm_w<true, 1, 8, 32, true, false, 0, true, HID, HID, 0>(
      Hh, nullptr, ws + OFF_W11, b11, nullptr, S, nullptr);
  __syncthreads();

  // P6: h = relu(t1 @ W12^T + b12) -> Hh f16
  gemm_w<false, 4, 8, HID, true, false, 0, false, HID, HID, 0>(
      nullptr, S, ws + OFF_W12, b12, nullptr, nullptr, Hh);
  __syncthreads();

  // P7: h + m^T@h -> S f32   [C = h folded in]
  gemm_aggr<8, HID, HID, true, HID>(Mh, Hh, Hh, S, nullptr);
  __syncthreads();

  // P8: relu((h+aggr) @ W21^T + b21) -> Hh
  gemm_w<false, 4, 8, HID, true, false, 0, false, HID, HID, 0>(
      nullptr, S, ws + OFF_W21, b21, nullptr, nullptr, Hh);
  __syncthreads();

  // P9: t = relu(... @ W22^T + b22) -> S (transformer residual stream)
  gemm_w<true, 4, 8, HID, true, false, 0, true, HID, HID, 0>(
      Hh, nullptr, ws + OFF_W22, b22, nullptr, S, nullptr);
  __syncthreads();

  // Transformer layers (V-projection only "attention")
#pragma unroll 1
  for (int l = 0; l < 2; ++l) {
    const _Float16* Wvf  = ws + OFF_WV  + (size_t)l * FR(4, 8);
    const _Float16* Wof  = ws + OFF_WO  + (size_t)l * FR(4, 8);
    const _Float16* Wf1f = ws + OFF_WF1 + (size_t)l * FR(4, 8);
    const _Float16* Wf2f = ws + OFF_WF2 + (size_t)l * FR(4, 8);
    const float* bv = bqkv + (size_t)l * 3 * HID + 2 * HID;

    // Qv = t @ Wv^T + bv -> Hh
    gemm_w<false, 4, 8, HID, false, false, 0, false, HID, HID, 0>(
        nullptr, S, Wvf, bv, nullptr, nullptr, Hh);
    __syncthreads();
    // t = t + Qv @ Wo^T + bo   (C preloaded from S; in-place safe per-tile)
    gemm_w<true, 4, 8, HID, false, true, HID, true, HID, HID, 0>(
        Hh, nullptr, Wof, bo + (size_t)l * HID, S, S, nullptr);
    __syncthreads();
    layernorm(S, ln1g + (size_t)l * HID, ln1b + (size_t)l * HID, stats, tid);
    __syncthreads();
    // ff1 = relu(t @ Wf1^T + bf1) -> Hh
    gemm_w<false, 4, 8, HID, true, false, 0, false, HID, HID, 0>(
        nullptr, S, Wf1f, bf1 + (size_t)l * HID, nullptr, nullptr, Hh);
    __syncthreads();
    // t = t + ff1 @ Wf2^T + bf2
    gemm_w<true, 4, 8, HID, false, true, HID, true, HID, HID, 0>(
        Hh, nullptr, Wf2f, bf2 + (size_t)l * HID, S, S, nullptr);
    __syncthreads();
    layernorm(S, ln2g + (size_t)l * HID, ln2b + (size_t)l * HID, stats, tid);
    __syncthreads();
  }

  // Final projection 128 -> 1
  if (tid < NAG) {
    float acc = bfc[0];
    const float* r = S + (size_t)tid * HID;
#pragma unroll 4
    for (int k = 0; k < HID; ++k) acc += r[k] * Wfc[k];
    out[(size_t)b * NAG + tid] = acc;
  }
}

// =====================================================================
extern "C" void kernel_launch(void* const* d_in, const int* in_sizes, int n_in,
                              void* d_out, int out_size, void* d_ws, size_t ws_size,
                              hipStream_t stream) {
  (void)in_sizes; (void)n_in; (void)out_size; (void)ws_size;
  const float* data = (const float*)d_in[0];
  const float* W_l1 = (const float*)d_in[1];  const float* b_l1 = (const float*)d_in[2];
  const float* W_l2 = (const float*)d_in[3];  const float* b_l2 = (const float*)d_in[4];
  const float* W11  = (const float*)d_in[5];  const float* b11  = (const float*)d_in[6];
  const float* W12  = (const float*)d_in[7];  const float* b12  = (const float*)d_in[8];
  const float* W21  = (const float*)d_in[9];  const float* b21  = (const float*)d_in[10];
  const float* W22  = (const float*)d_in[11]; const float* b22  = (const float*)d_in[12];
  const float* Wqkv = (const float*)d_in[13]; const float* bqkv = (const float*)d_in[14];
  const float* Wo   = (const float*)d_in[15]; const float* bo   = (const float*)d_in[16];
  const float* ln1g = (const float*)d_in[17]; const float* ln1b = (const float*)d_in[18];
  const float* Wf1  = (const float*)d_in[19]; const float* bf1  = (const float*)d_in[20];
  const float* Wf2  = (const float*)d_in[21]; const float* bf2  = (const float*)d_in[22];
  const float* ln2g = (const float*)d_in[23]; const float* ln2b = (const float*)d_in[24];
  const float* Wfc  = (const float*)d_in[25]; const float* bfc  = (const float*)d_in[26];

  _Float16* ws = (_Float16*)d_ws;

  auto conv = [&](const float* W, size_t off, int K, int N, int KT, int NT) {
    int total = KT * NT * 32;
    convert_w<<<(total + 255) / 256, 256, 0, stream>>>(W, ws + off, K, N, KT, NT);
  };
  conv(W_l1, OFF_WL1, LIDAR, HID, 1, 8);
  conv(W_l2, OFF_WL2, HID, PROC, 4, 2);
  conv(W11,  OFF_W11, FEAT, HID, 1, 8);
  conv(W12,  OFF_W12, HID, HID, 4, 8);
  conv(W21,  OFF_W21, HID, HID, 4, 8);
  conv(W22,  OFF_W22, HID, HID, 4, 8);
  for (int l = 0; l < 2; ++l) {
    conv(Wqkv + (size_t)l * 3 * HID * HID + 2 * HID * HID, OFF_WV + (size_t)l * FR(4, 8),
         HID, HID, 4, 8);
    conv(Wo  + (size_t)l * HID * HID, OFF_WO  + (size_t)l * FR(4, 8), HID, HID, 4, 8);
    conv(Wf1 + (size_t)l * HID * HID, OFF_WF1 + (size_t)l * FR(4, 8), HID, HID, 4, 8);
    conv(Wf2 + (size_t)l * HID * HID, OFF_WF2 + (size_t)l * FR(4, 8), HID, HID, 4, 8);
  }

  gin_critic_fused<<<BATCH, 256, 0, stream>>>(
      data, b_l1, b_l2, b11, b12, b21, b22, bqkv, bo,
      ln1g, ln1b, bf1, bf2, ln2g, ln2b, Wfc, bfc, ws, (float*)d_out);
}
